// LSTMmedGANwatt_36404142801114
// MI455X (gfx1250) — matrix operand order repaired and checked by reading, not compiled, not run on hardware
//
#include <hip/hip_runtime.h>
#include <hip/hip_bf16.h>

typedef __attribute__((ext_vector_type(16))) _Float16 v16h;
typedef __attribute__((ext_vector_type(8)))  _Float16 v8h;
typedef __attribute__((ext_vector_type(8)))  float    v8f;

#define B_ 128
#define V_ 64
#define S_ 40
#define D_ 256
#define H_ 256

__device__ __forceinline__ float sigmoidf_(float x) { return 1.0f / (1.0f + __expf(-x)); }

// A-tile loader: A is MxK row-major f16. contiguous-K A 16x32 layout (same order as load_b16):
// lanes 0-15 (g=0): K = k..k+15; lanes 16-31 (g=1): K = k+16..k+31
__device__ __forceinline__ v16h load_a16(const _Float16* __restrict__ A, int lda,
                                         int m0, int k, int lane) {
    int m = m0 + (lane & 15);
    int g = lane >> 4;
    return *(const v16h*)(A + (size_t)m * lda + k + g * 16);
}

// B-tile loader: W is NxK row-major f16 (we compute A @ W^T). B KxN layout:
// lane%16 selects N column, lane group selects K half: 16 contiguous K per lane.
__device__ __forceinline__ v16h load_b16(const _Float16* __restrict__ W, int ldw,
                                         int n0, int k, int lane) {
    int n = n0 + (lane & 15);
    int g = lane >> 4;
    return *(const v16h*)(W + (size_t)n * ldw + k + g * 16);
}

// ---------------- embedding gather: v = relu(emb[seqs]).sum(-2) -----------------
__global__ __launch_bounds__(256) void embed_kernel(const int* __restrict__ seqs,
                                                    const float* __restrict__ emb,
                                                    float* __restrict__ v_f,
                                                    _Float16* __restrict__ v_h) {
    int r = blockIdx.x;        // b*V + vi, 8192 rows
    int d = threadIdx.x;       // 256
    __shared__ int idxs[S_];
    if (d < S_) idxs[d] = seqs[(size_t)r * S_ + d];
    __syncthreads();
    float acc = 0.0f;
#pragma unroll 4
    for (int s = 0; s < S_; s++) {
        float e = emb[(size_t)idxs[s] * D_ + d];
        acc += fmaxf(e, 0.0f);
    }
    size_t o = (size_t)r * D_ + d;
    v_f[o] = acc;
    v_h[o] = (_Float16)acc;
}

// ---------------- generic WMMA GEMM: C[MxN] = act(A[MxK] @ W[NxK]^T + bias) -----
// grid = (M/16, N/128), block = 256 (8 waves), wave -> one 16x16 tile.
__global__ __launch_bounds__(256) void gemm_wmma_kernel(const _Float16* __restrict__ A,
                                                        const _Float16* __restrict__ W,
                                                        const float* __restrict__ bias,
                                                        float* __restrict__ Cf,
                                                        _Float16* __restrict__ Ch,
                                                        int N, int K, int act) {
    int lane = threadIdx.x & 31;
    int wave = threadIdx.x >> 5;
    int m0 = blockIdx.x * 16;
    int n0 = (blockIdx.y * 8 + wave) * 16;
    int n  = n0 + (lane & 15);
    float bv = bias ? bias[n] : 0.0f;
    v8f acc;
#pragma unroll
    for (int i = 0; i < 8; i++) acc[i] = bv;
    for (int k = 0; k < K; k += 32) {
        v16h a = load_a16(A, K, m0, k, lane);
        v16h b = load_b16(W, K, n0, k, lane);
        acc = __builtin_amdgcn_wmma_f32_16x16x32_f16(false, a, false, b,
                                                     (short)0, acc, false, false);
    }
    int mb = (lane >> 4) * 8;   // C layout: lanes 0-15 -> M=0..7, lanes 16-31 -> M=8..15
#pragma unroll
    for (int i = 0; i < 8; i++) {
        int row = m0 + mb + i;
        float v = acc[i];
        if (act == 1) v = fmaxf(v, 0.0f);
        else if (act == 2) v = tanhf(v);
        size_t idx = (size_t)row * N + n;
        if (Cf) Cf[idx] = v;
        if (Ch) Ch[idx] = (_Float16)v;
    }
}

// ---------------- fused LSTM step: gates = xg_t + h @ Whh^T, then pointwise -----
// grid = 32 blocks x 128 threads (4 waves); 128 waves total = 8 Mtiles x 16 Ntiles.
// Each wave computes i/f/g/o 16x16 tiles at gate offsets 0/H/2H/3H, then updates c,h.
__global__ __launch_bounds__(128) void lstm_step_kernel(const float* __restrict__ xg,
                                                        const _Float16* __restrict__ Whh,
                                                        const _Float16* __restrict__ h_prev,
                                                        float* __restrict__ c,
                                                        _Float16* __restrict__ hs_h,
                                                        _Float16* __restrict__ h_next,
                                                        int t) {
    int lane = threadIdx.x & 31;
    int wave = threadIdx.x >> 5;
    int wg = blockIdx.x * 4 + wave;       // 0..127
    int m0 = (wg >> 4) * 16;              // batch tile
    int n0 = (wg & 15) * 16;              // column tile within H
    int n  = n0 + (lane & 15);
    int mb = (lane >> 4) * 8;
    v8f ai, af, ag, ao;
#pragma unroll
    for (int i = 0; i < 8; i++) {
        int b = m0 + mb + i;
        const float* xr = xg + (size_t)(b * V_ + t) * (4 * H_);
        ai[i] = xr[0 * H_ + n];
        af[i] = xr[1 * H_ + n];
        ag[i] = xr[2 * H_ + n];
        ao[i] = xr[3 * H_ + n];
    }
    for (int k = 0; k < H_; k += 32) {
        v16h a  = load_a16(h_prev, H_, m0, k, lane);
        v16h bi = load_b16(Whh, H_, 0 * H_ + n0, k, lane);
        v16h bf = load_b16(Whh, H_, 1 * H_ + n0, k, lane);
        v16h bg = load_b16(Whh, H_, 2 * H_ + n0, k, lane);
        v16h bo = load_b16(Whh, H_, 3 * H_ + n0, k, lane);
        ai = __builtin_amdgcn_wmma_f32_16x16x32_f16(false, a, false, bi, (short)0, ai, false, false);
        af = __builtin_amdgcn_wmma_f32_16x16x32_f16(false, a, false, bf, (short)0, af, false, false);
        ag = __builtin_amdgcn_wmma_f32_16x16x32_f16(false, a, false, bg, (short)0, ag, false, false);
        ao = __builtin_amdgcn_wmma_f32_16x16x32_f16(false, a, false, bo, (short)0, ao, false, false);
    }
#pragma unroll
    for (int i = 0; i < 8; i++) {
        int b = m0 + mb + i;
        float iv = sigmoidf_(ai[i]);
        float fv = sigmoidf_(af[i]);
        float gv = tanhf(ag[i]);
        float ov = sigmoidf_(ao[i]);
        size_t ci = (size_t)b * H_ + n;
        float cv = fv * c[ci] + iv * gv;
        c[ci] = cv;
        float hv = ov * tanhf(cv);
        _Float16 hh = (_Float16)hv;
        hs_h[(size_t)(b * V_ + t) * H_ + n] = hh;
        h_next[ci] = hh;
    }
}

// ---------------- attention + bar_v construction --------------------------------
// block per (b, ti): ti==0 copies v[:,0,:]; else attention over [e_k, wh[b,ti-1]]
__global__ __launch_bounds__(256) void attention_bar_kernel(const float* __restrict__ v,
                                                            const float* __restrict__ wh,
                                                            const float* __restrict__ W1,
                                                            const float* __restrict__ b1,
                                                            const float* __restrict__ W2,
                                                            const float* __restrict__ b2,
                                                            float* __restrict__ bar_f,
                                                            _Float16* __restrict__ bar_h) {
    int blk = blockIdx.x;
    int b = blk >> 6;
    int ti = blk & 63;
    int tid = threadIdx.x;   // 256
    const float* ek = v + (size_t)(b * V_) * D_;
    size_t orow = (size_t)(b * V_ + ti) * D_;
    __shared__ float cat[2 * D_];
    __shared__ float part[256];
    __shared__ float a1[64];
    __shared__ float probs[2];
    if (ti == 0) {
        float val = ek[tid];
        bar_f[orow + tid] = val;
        bar_h[orow + tid] = (_Float16)val;
        return;
    }
    int t = ti - 1;
    const float* whp = wh + (size_t)(b * V_ + t) * D_;
    cat[tid] = ek[tid];
    cat[D_ + tid] = whp[tid];
    __syncthreads();
    int j = tid >> 2, p4 = tid & 3;
    {
        const float* wrow = W1 + (size_t)j * (2 * D_) + p4 * 128;
        const float* crow = cat + p4 * 128;
        float s = 0.0f;
#pragma unroll 4
        for (int x = 0; x < 128; x++) s += crow[x] * wrow[x];
        part[tid] = s;
    }
    __syncthreads();
    if (p4 == 0) a1[j] = tanhf(part[tid] + part[tid + 1] + part[tid + 2] + part[tid + 3] + b1[j]);
    __syncthreads();
    if (tid < 2) {
        float l = b2[tid];
        for (int jj = 0; jj < 64; jj++) l += a1[jj] * W2[tid * 64 + jj];
        part[tid] = l;
    }
    __syncthreads();
    if (tid == 0) {
        float l0 = part[0], l1 = part[1];
        float m = fmaxf(l0, l1);
        float e0 = __expf(l0 - m), e1 = __expf(l1 - m);
        float inv = 1.0f / (e0 + e1);
        probs[0] = e0 * inv;
        probs[1] = e1 * inv;
    }
    __syncthreads();
    float bv = ek[tid] * probs[0] + whp[tid] * probs[1];
    bar_f[orow + tid] = bv;
    bar_h[orow + tid] = (_Float16)bv;
}

// ---------------- final 2-wide classifier head ----------------------------------
__global__ __launch_bounds__(256) void final_out_kernel(const float* __restrict__ f2,
                                                        const float* __restrict__ CoW,
                                                        const float* __restrict__ Cob,
                                                        float* __restrict__ out) {
    int tid = threadIdx.x;      // 256 = 128 batches x 2 classes
    int b = tid >> 1, p = tid & 1;
    float s = Cob[p];
    const float* row = f2 + (size_t)b * 512;
    const float* w = CoW + (size_t)p * 512;
#pragma unroll 4
    for (int x = 0; x < 512; x++) s += row[x] * w[x];
    out[b * 2 + p] = s;
}

// ---------------- small helpers --------------------------------------------------
__global__ void f32_to_f16_kernel(const float* __restrict__ s, _Float16* __restrict__ d, int n) {
    int i = blockIdx.x * blockDim.x + threadIdx.x;
    if (i < n) d[i] = (_Float16)s[i];
}
__global__ void add_bias_kernel(const float* a, const float* b, float* o, int n) {
    int i = blockIdx.x * blockDim.x + threadIdx.x;
    if (i < n) o[i] = a[i] + b[i];
}
__global__ void copy2_kernel(const float* __restrict__ s, float* __restrict__ d1,
                             float* __restrict__ d2, int n) {
    int i = blockIdx.x * blockDim.x + threadIdx.x;
    if (i < n) { float v = s[i]; d1[i] = v; d2[i] = v; }
}
__global__ void copy1_kernel(const float* __restrict__ s, float* __restrict__ d, int n) {
    int i = blockIdx.x * blockDim.x + threadIdx.x;
    if (i < n) d[i] = s[i];
}
__global__ void i2f_kernel(const int* __restrict__ s, float* __restrict__ d, int n) {
    int i = blockIdx.x * blockDim.x + threadIdx.x;
    if (i < n) d[i] = (float)s[i];
}

extern "C" void kernel_launch(void* const* d_in, const int* in_sizes, int n_in,
                              void* d_out, int out_size, void* d_ws, size_t ws_size,
                              hipStream_t stream) {
    (void)in_sizes; (void)n_in; (void)out_size; (void)ws_size;
    const int*   input_seqs = (const int*)d_in[0];
    const float* seq_time   = (const float*)d_in[3];
    const int*   label      = (const int*)d_in[5];
    const float* emb        = (const float*)d_in[6];
    const float* W_ih       = (const float*)d_in[7];
    const float* W_hh       = (const float*)d_in[8];
    const float* b_ih       = (const float*)d_in[9];
    const float* b_hh       = (const float*)d_in[10];
    const float* Whk        = (const float*)d_in[11];
    const float* bhk        = (const float*)d_in[12];
    const float* W1         = (const float*)d_in[13];
    const float* b1         = (const float*)d_in[14];
    const float* W2         = (const float*)d_in[15];
    const float* b2         = (const float*)d_in[16];
    const float* C1W        = (const float*)d_in[17];
    const float* C1b        = (const float*)d_in[18];
    const float* C2W        = (const float*)d_in[19];
    const float* C2b        = (const float*)d_in[20];
    const float* CoW        = (const float*)d_in[21];
    const float* Cob        = (const float*)d_in[22];
    float* out = (float*)d_out;

    // workspace carve-out (256B aligned)
    char* ws = (char*)d_ws;
    size_t off = 0;
    auto alloc = [&](size_t bytes) -> char* {
        off = (off + 255) & ~(size_t)255;
        char* p = ws + off;
        off += bytes;
        return p;
    };
    _Float16* Wih_h   = (_Float16*)alloc((size_t)1024 * 256 * 2);
    _Float16* Whh_h   = (_Float16*)alloc((size_t)1024 * 256 * 2);
    _Float16* Whk_h   = (_Float16*)alloc((size_t)256 * 256 * 2);
    _Float16* C1W_h   = (_Float16*)alloc((size_t)1024 * 256 * 2);
    _Float16* C2W_h   = (_Float16*)alloc((size_t)512 * 1024 * 2);
    float*    bias_cmb= (float*)   alloc((size_t)1024 * 4);
    float*    v_f     = (float*)   alloc((size_t)B_ * V_ * D_ * 4);
    _Float16* v_h     = (_Float16*)alloc((size_t)B_ * V_ * D_ * 2);
    float*    xg      = (float*)   alloc((size_t)B_ * V_ * 4 * H_ * 4); // reused by both LSTMs
    _Float16* hs_h    = (_Float16*)alloc((size_t)B_ * V_ * H_ * 2);     // reused
    float*    wh_f    = (float*)   alloc((size_t)B_ * V_ * D_ * 4);
    float*    bar_f   = (float*)   alloc((size_t)B_ * V_ * D_ * 4);
    _Float16* bar_h   = (_Float16*)alloc((size_t)B_ * V_ * D_ * 2);
    float*    c_buf   = (float*)   alloc((size_t)B_ * H_ * 4);
    _Float16* hA      = (_Float16*)alloc((size_t)B_ * H_ * 2);
    _Float16* hB      = (_Float16*)alloc((size_t)B_ * H_ * 2);
    _Float16* f1_h    = (_Float16*)alloc((size_t)B_ * 1024 * 2);
    float*    f2_f    = (float*)   alloc((size_t)B_ * 512 * 4);

    // output layout (flat, return order)
    float* out_og   = out;                      // 128*2
    float* out_fake = out + 256;                // 128*2
    float* out_dec  = out + 512;                // 2097152
    float* out_gen  = out + 512 + 2097152;      // 2097152
    float* out_time = out + 512 + 2 * 2097152;  // 8192
    float* out_lab  = out_time + 8192;          // 128

    // weight conversions to f16
    f32_to_f16_kernel<<<1024, 256, 0, stream>>>(W_ih, Wih_h, 262144);
    f32_to_f16_kernel<<<1024, 256, 0, stream>>>(W_hh, Whh_h, 262144);
    f32_to_f16_kernel<<<256, 256, 0, stream>>>(Whk, Whk_h, 65536);
    f32_to_f16_kernel<<<1024, 256, 0, stream>>>(C1W, C1W_h, 262144);
    f32_to_f16_kernel<<<2048, 256, 0, stream>>>(C2W, C2W_h, 524288);
    add_bias_kernel<<<4, 256, 0, stream>>>(b_ih, b_hh, bias_cmb, 1024);

    // v = relu(emb[seqs]).sum(-2)
    embed_kernel<<<B_ * V_, 256, 0, stream>>>(input_seqs, emb, v_f, v_h);

    // xg1 = v @ W_ih^T + (b_ih + b_hh)   M=8192 N=1024 K=256
    gemm_wmma_kernel<<<dim3(512, 8), 256, 0, stream>>>(v_h, Wih_h, bias_cmb, xg, nullptr, 1024, 256, 0);

    // LSTM 1
    (void)hipMemsetAsync(c_buf, 0, (size_t)B_ * H_ * 4, stream);
    (void)hipMemsetAsync(hA, 0, (size_t)B_ * H_ * 2, stream);
    _Float16 *cur = hA, *nxt = hB;
    for (int t = 0; t < V_; t++) {
        lstm_step_kernel<<<32, 128, 0, stream>>>(xg, Whh_h, cur, c_buf, hs_h, nxt, t);
        _Float16* tmp = cur; cur = nxt; nxt = tmp;
    }

    // final_og = classify(h_last1)
    gemm_wmma_kernel<<<dim3(8, 8), 256, 0, stream>>>(cur, C1W_h, C1b, nullptr, f1_h, 1024, 256, 1);
    gemm_wmma_kernel<<<dim3(8, 4), 256, 0, stream>>>(f1_h, C2W_h, C2b, f2_f, nullptr, 512, 1024, 1);
    final_out_kernel<<<1, 256, 0, stream>>>(f2_f, CoW, Cob, out_og);

    // wh = hs1 @ Whk^T + bhk   M=8192 N=256 K=256
    gemm_wmma_kernel<<<dim3(512, 2), 256, 0, stream>>>(hs_h, Whk_h, bhk, wh_f, nullptr, 256, 256, 0);

    // attention -> bar_v
    attention_bar_kernel<<<B_ * V_, 256, 0, stream>>>(v_f, wh_f, W1, b1, W2, b2, bar_f, bar_h);

    // xg2 = bar_v @ W_ih^T + biases (reuse xg)
    gemm_wmma_kernel<<<dim3(512, 8), 256, 0, stream>>>(bar_h, Wih_h, bias_cmb, xg, nullptr, 1024, 256, 0);

    // LSTM 2
    (void)hipMemsetAsync(c_buf, 0, (size_t)B_ * H_ * 4, stream);
    (void)hipMemsetAsync(hA, 0, (size_t)B_ * H_ * 2, stream);
    cur = hA; nxt = hB;
    for (int t = 0; t < V_; t++) {
        lstm_step_kernel<<<32, 128, 0, stream>>>(xg, Whh_h, cur, c_buf, hs_h, nxt, t);
        _Float16* tmp = cur; cur = nxt; nxt = tmp;
    }

    // final_fake = classify(h_last2)
    gemm_wmma_kernel<<<dim3(8, 8), 256, 0, stream>>>(cur, C1W_h, C1b, nullptr, f1_h, 1024, 256, 1);
    gemm_wmma_kernel<<<dim3(8, 4), 256, 0, stream>>>(f1_h, C2W_h, C2b, f2_f, nullptr, 512, 1024, 1);
    final_out_kernel<<<1, 256, 0, stream>>>(f2_f, CoW, Cob, out_fake);

    // remaining outputs: Dec_V = Gen_V = bar_v, seq_time_step, label
    copy2_kernel<<<8192, 256, 0, stream>>>(bar_f, out_dec, out_gen, B_ * V_ * D_);
    copy1_kernel<<<32, 256, 0, stream>>>(seq_time, out_time, B_ * V_);
    i2f_kernel<<<1, 128, 0, stream>>>(label, out_lab, B_);
}
